// Attention_19954418057485
// MI455X (gfx1250) — compile-verified, matrix operand order checked by reading, not run on hardware
//
#include <hip/hip_runtime.h>

// B=2, N=2048, C=1024, H=16, D=64
#define BB 2
#define NN 2048
#define CC 1024
#define HH 16
#define DD 64

typedef __attribute__((ext_vector_type(16))) _Float16 v16h;
typedef __attribute__((ext_vector_type(8)))  _Float16 v8h;
typedef __attribute__((ext_vector_type(4)))  _Float16 v4h;
typedef __attribute__((ext_vector_type(8)))  float    v8f;
typedef int v4i_gcc __attribute__((vector_size(4 * sizeof(int))));

#if __has_builtin(__builtin_amdgcn_global_load_async_to_lds_b128)
#define HAVE_ASYNC_LDS 1
#else
#define HAVE_ASYNC_LDS 0
#endif

// Copy 16 bytes global -> LDS. Prefer the CDNA5 async-DMA path (ASYNCcnt).
// Builtin signature (from compiler diagnostic): (v4i AS1*, v4i AS3*, Ii, Ii).
__device__ __forceinline__ void async_copy16(const _Float16* g, _Float16* l) {
#if HAVE_ASYNC_LDS
  __builtin_amdgcn_global_load_async_to_lds_b128(
      (__attribute__((address_space(1))) v4i_gcc*)(const void*)g,
      (__attribute__((address_space(3))) v4i_gcc*)(void*)l, 0, 0);
#else
  *(v8h*)l = *(const v8h*)g;
#endif
}

__device__ __forceinline__ void wait_async_lds() {
#if HAVE_ASYNC_LDS
  asm volatile("s_wait_asynccnt 0x0" ::: "memory");
#endif
}

__device__ __forceinline__ v8f wmma_f16(v16h a, v16h b, v8f c) {
  // v_wmma_f32_16x16x32_f16 : D = A*B + C
  return __builtin_amdgcn_wmma_f32_16x16x32_f16(false, a, false, b, (short)0, c, false, false);
}

// Build a 16x32 f16 A/B fragment for one lane (ISA 7.12.2 layout):
// lanes 0-15: VGPR0-3 = K0..7, VGPR4-7 = K16..23; lanes 16-31: K8..15, K24..31.
__device__ __forceinline__ v16h frag16(const _Float16* p, int halfid) {
  v8h lo = *(const v8h*)(p + halfid * 8);
  v8h hi = *(const v8h*)(p + 16 + halfid * 8);
  v16h r;
#pragma unroll
  for (int i = 0; i < 8; ++i) { r[i] = lo[i]; r[i + 8] = hi[i]; }
  return r;
}

__device__ __forceinline__ float wave16_max(float v) {
#pragma unroll
  for (int m = 1; m <= 8; m <<= 1) v = fmaxf(v, __shfl_xor(v, m, 32));
  return v;
}
__device__ __forceinline__ float wave16_sum(float v) {
#pragma unroll
  for (int m = 1; m <= 8; m <<= 1) v += __shfl_xor(v, m, 32);
  return v;
}

// ---------------------------------------------------------------------------
// Kernel 1: qkv = x @ w_qkv^T  (M=4096, N=3072, K=1024), f16 WMMA.
// Block = 256 thr (8 waves) computes 128(M) x 128(N); wave tile = 32x64.
// f32 -> f16 conversion happens while staging into LDS (must go through VALU,
// so no async DMA here). Results scattered into q/k/v as [B,H,N,D] f32.
// ---------------------------------------------------------------------------
__global__ __launch_bounds__(256) void qkv_gemm(const float* __restrict__ x,
                                                const float* __restrict__ w,
                                                float* __restrict__ qf,
                                                float* __restrict__ kf,
                                                float* __restrict__ vf) {
  __shared__ _Float16 Alds[128 * 32];
  __shared__ _Float16 Blds[128 * 32];

  const int tid = threadIdx.x;
  const int wave = tid >> 5, lane = tid & 31;
  const int halfid = lane >> 4, ln16 = lane & 15;
  const int nt = blockIdx.x % 24, mt = blockIdx.x / 24;
  const int m0 = mt * 128, n0 = nt * 128;
  const int wm = wave >> 1, wn = wave & 1;

  v8f acc[2][4];
#pragma unroll
  for (int i = 0; i < 2; ++i)
#pragma unroll
    for (int j = 0; j < 4; ++j)
#pragma unroll
      for (int r = 0; r < 8; ++r) acc[i][j][r] = 0.0f;

  for (int k0 = 0; k0 < CC; k0 += 32) {
    // stage A: 128 rows x 32 k (f32 -> f16)
#pragma unroll
    for (int i = 0; i < 4; ++i) {
      int id = tid + i * 256;                 // 0..1023
      int row = id >> 3, kc = (id & 7) * 4;
      float4 f = *(const float4*)(x + (size_t)(m0 + row) * CC + k0 + kc);
      v4h hh;
      hh[0] = (_Float16)f.x; hh[1] = (_Float16)f.y;
      hh[2] = (_Float16)f.z; hh[3] = (_Float16)f.w;
      *(v4h*)&Alds[row * 32 + kc] = hh;
    }
    // stage B: 128 out-channels x 32 k (f32 -> f16)
#pragma unroll
    for (int i = 0; i < 4; ++i) {
      int id = tid + i * 256;
      int row = id >> 3, kc = (id & 7) * 4;
      const float* src = w + (size_t)(n0 + row) * CC + k0 + kc;
      float4 f = *(const float4*)src;
      v4h hh;
      hh[0] = (_Float16)f.x; hh[1] = (_Float16)f.y;
      hh[2] = (_Float16)f.z; hh[3] = (_Float16)f.w;
      *(v4h*)&Blds[row * 32 + kc] = hh;
      if (i == 0) __builtin_prefetch(src + 32, 0, 1);   // next k-step
    }
    __syncthreads();

    v16h a0 = frag16(&Alds[(wm * 32 + ln16) * 32], halfid);
    v16h a1 = frag16(&Alds[(wm * 32 + 16 + ln16) * 32], halfid);
#pragma unroll
    for (int j = 0; j < 4; ++j) {
      v16h bj = frag16(&Blds[(wn * 64 + j * 16 + ln16) * 32], halfid);
      acc[0][j] = wmma_f16(a0, bj, acc[0][j]);
      acc[1][j] = wmma_f16(a1, bj, acc[1][j]);
    }
    __syncthreads();
  }

  // scatter into q/k/v [B,H,N,D]; this block's 128 channels stay inside one
  // of q/k/v (boundaries at multiples of 1024).
  const int which = n0 >> 10;
  float* dst = (which == 0) ? qf : ((which == 1) ? kf : vf);
  const int bsel = m0 >> 11;
  const int nbase = m0 & (NN - 1);
#pragma unroll
  for (int mi = 0; mi < 2; ++mi)
#pragma unroll
    for (int ni = 0; ni < 4; ++ni) {
      int o = n0 + wn * 64 + ni * 16 + ln16;
      int h = (o >> 6) & (HH - 1);
      int d = o & (DD - 1);
#pragma unroll
      for (int r = 0; r < 8; ++r) {
        int row_local = wm * 32 + mi * 16 + halfid * 8 + r;
        size_t idx = (((size_t)bsel * HH + h) * NN + nbase + row_local) * DD + d;
        dst[idx] = acc[mi][ni][r];
      }
    }
}

// ---------------------------------------------------------------------------
// Kernel 2: per-head LayerNorm + RoPE (+ q *= D^-1/2), f32 -> f16.
// One wave per [B,H,N] row of 64; lane holds elems (d, d+32).
// ---------------------------------------------------------------------------
__global__ __launch_bounds__(256) void lnrope(const float* __restrict__ qf,
                                              const float* __restrict__ kf,
                                              const float* __restrict__ vf,
                                              const float* __restrict__ fcos,
                                              const float* __restrict__ fsin,
                                              _Float16* __restrict__ qh,
                                              _Float16* __restrict__ kh,
                                              _Float16* __restrict__ vh) {
  const int wave = threadIdx.x >> 5, lane = threadIdx.x & 31;
  const int row = blockIdx.x * 8 + wave;       // 0 .. B*H*N-1
  const int n = row & (NN - 1);
  const size_t base = (size_t)row * DD;

  float q0 = qf[base + lane], q1 = qf[base + lane + 32];
  float k0 = kf[base + lane], k1 = kf[base + lane + 32];
  float v0 = vf[base + lane], v1 = vf[base + lane + 32];

  float qm = wave16_sum(q0 + q1);
  qm += __shfl_xor(qm, 16, 32);  qm *= (1.0f / 64.0f);
  float qv = wave16_sum((q0 - qm) * (q0 - qm) + (q1 - qm) * (q1 - qm));
  qv += __shfl_xor(qv, 16, 32);  qv *= (1.0f / 64.0f);
  float qs = __frsqrt_rn(qv + 1e-6f);
  q0 = (q0 - qm) * qs; q1 = (q1 - qm) * qs;

  float km = wave16_sum(k0 + k1);
  km += __shfl_xor(km, 16, 32);  km *= (1.0f / 64.0f);
  float kv = wave16_sum((k0 - km) * (k0 - km) + (k1 - km) * (k1 - km));
  kv += __shfl_xor(kv, 16, 32);  kv *= (1.0f / 64.0f);
  float ks = __frsqrt_rn(kv + 1e-6f);
  k0 = (k0 - km) * ks; k1 = (k1 - km) * ks;

  const float c0 = fcos[n * DD + lane], c1 = fcos[n * DD + lane + 32];
  const float s0 = fsin[n * DD + lane], s1 = fsin[n * DD + lane + 32];
  float qr0 = q0 * c0 - q1 * s0, qr1 = q1 * c1 + q0 * s1;
  float kr0 = k0 * c0 - k1 * s0, kr1 = k1 * c1 + k0 * s1;

  const float scale = 0.125f;                 // D^-1/2 folded into q
  qh[base + lane]      = (_Float16)(qr0 * scale);
  qh[base + lane + 32] = (_Float16)(qr1 * scale);
  kh[base + lane]      = (_Float16)kr0;
  kh[base + lane + 32] = (_Float16)kr1;
  vh[base + lane]      = (_Float16)v0;
  vh[base + lane + 32] = (_Float16)v1;
}

// ---------------------------------------------------------------------------
// Kernel 3: flash attention per (b,h, 128-query tile). 8 waves x 16 queries.
// Key chunks of 64; K staged [key][d] via async global->LDS DMA, V staged
// transposed [d][key] through VGPRs.
// ---------------------------------------------------------------------------
__global__ __launch_bounds__(256) void attn(const _Float16* __restrict__ qh,
                                            const _Float16* __restrict__ kh,
                                            const _Float16* __restrict__ vh,
                                            _Float16* __restrict__ attnw) {
  __shared__ _Float16 Klds[64 * 64];          // [key][d]
  __shared__ _Float16 Vlds[64 * 64];          // [d][key]
  __shared__ _Float16 Plds[8 * 16 * 64];      // per-wave P scratch

  const int tid = threadIdx.x;
  const int wave = tid >> 5, lane = tid & 31;
  const int halfid = lane >> 4, ln16 = lane & 15;
  const int bh = blockIdx.x >> 4;             // 0..31
  const int qt = blockIdx.x & 15;
  const int b = bh >> 4, h = bh & 15;

  const _Float16* qrow = qh + ((size_t)bh * NN + qt * 128 + wave * 16 + ln16) * DD;
  const v16h qa0 = frag16(qrow, halfid);
  const v16h qa1 = frag16(qrow + 32, halfid);

  v8f oacc[4];
#pragma unroll
  for (int dt = 0; dt < 4; ++dt)
#pragma unroll
    for (int r = 0; r < 8; ++r) oacc[dt][r] = 0.0f;
  float mrow[8], lrow[8];
#pragma unroll
  for (int r = 0; r < 8; ++r) { mrow[r] = -1e30f; lrow[r] = 0.0f; }

  const _Float16* kgb = kh + (size_t)bh * NN * DD;
  const _Float16* vgb = vh + (size_t)bh * NN * DD;

  for (int kc = 0; kc < NN / 64; ++kc) {
    // stage K chunk: flat 64x64 halves via async DMA (2x b128 per thread)
    {
      const _Float16* src = kgb + kc * 64 * DD;
      async_copy16(src + tid * 16,     &Klds[tid * 16]);
      async_copy16(src + tid * 16 + 8, &Klds[tid * 16 + 8]);
    }
    // stage V chunk transposed: [d][key]
    {
      const _Float16* src = vgb + kc * 64 * DD;
#pragma unroll
      for (int i = 0; i < 2; ++i) {
        int c = tid + i * 256;                // 0..511
        int key = c >> 3, d8 = (c & 7) * 8;
        v8h vv = *(const v8h*)(src + key * DD + d8);
#pragma unroll
        for (int j = 0; j < 8; ++j) Vlds[(d8 + j) * 64 + key] = vv[j];
      }
    }
    wait_async_lds();
    __syncthreads();

    // S = Q K^T  (4 key sub-tiles of 16, contract over D=64)
    v8f sc[4];
#pragma unroll
    for (int kt = 0; kt < 4; ++kt) {
      const _Float16* kb = &Klds[(kt * 16 + ln16) * 64];
      v16h b0 = frag16(kb, halfid);
      v16h b1 = frag16(kb + 32, halfid);
      v8f s;
#pragma unroll
      for (int r = 0; r < 8; ++r) s[r] = 0.0f;
      s = wmma_f16(qa0, b0, s);
      s = wmma_f16(qa1, b1, s);
      sc[kt] = s;
    }

    // online softmax
    float mnew[8], scl[8];
#pragma unroll
    for (int r = 0; r < 8; ++r) {
      float m = fmaxf(fmaxf(sc[0][r], sc[1][r]), fmaxf(sc[2][r], sc[3][r]));
      m = wave16_max(m);
      mnew[r] = fmaxf(mrow[r], m);
      scl[r] = __expf(mrow[r] - mnew[r]);
    }
#pragma unroll
    for (int r = 0; r < 8; ++r) {
      float s4 = 0.0f;
#pragma unroll
      for (int kt = 0; kt < 4; ++kt) {
        float p = __expf(sc[kt][r] - mnew[r]);
        sc[kt][r] = p;
        s4 += p;
      }
      s4 = wave16_sum(s4);
      lrow[r] = lrow[r] * scl[r] + s4;
      mrow[r] = mnew[r];
    }
#pragma unroll
    for (int dt = 0; dt < 4; ++dt)
#pragma unroll
      for (int r = 0; r < 8; ++r) oacc[dt][r] *= scl[r];

    // P: C-layout regs -> A-fragment layout via per-wave LDS scratch
    _Float16* pw = &Plds[wave * 1024];
#pragma unroll
    for (int kt = 0; kt < 4; ++kt)
#pragma unroll
      for (int r = 0; r < 8; ++r)
        pw[(halfid * 8 + r) * 64 + kt * 16 + ln16] = (_Float16)sc[kt][r];
    asm volatile("s_wait_dscnt 0x0" ::: "memory");   // per-wave LDS RAW fence

    v16h pa0 = frag16(&pw[ln16 * 64], halfid);
    v16h pa1 = frag16(&pw[ln16 * 64 + 32], halfid);

    // O += P V  (4 d sub-tiles, contract over 64 keys)
#pragma unroll
    for (int dt = 0; dt < 4; ++dt) {
      const _Float16* vb = &Vlds[(dt * 16 + ln16) * 64];
      v16h vb0 = frag16(vb, halfid);
      v16h vb1 = frag16(vb + 32, halfid);
      oacc[dt] = wmma_f16(pa0, vb0, oacc[dt]);
      oacc[dt] = wmma_f16(pa1, vb1, oacc[dt]);
    }
    __syncthreads();
  }

  // normalize, emit f16 attn_out [B*N, C] (col = h*64 + d)
  const int qrow0 = qt * 128 + wave * 16 + halfid * 8;
#pragma unroll
  for (int r = 0; r < 8; ++r) {
    float inv = 1.0f / lrow[r];
#pragma unroll
    for (int dt = 0; dt < 4; ++dt) {
      size_t idx = ((size_t)b * NN + qrow0 + r) * CC + h * DD + dt * 16 + ln16;
      attnw[idx] = (_Float16)(oacc[dt][r] * inv);
    }
  }
}

// ---------------------------------------------------------------------------
// Kernel 4: out = attn_out @ w_proj^T + b_proj  (M=4096, N=1024, K=1024)
// Block 128x128, wave 32x64. A (f16) staged via async DMA; B converted f32->f16.
// ---------------------------------------------------------------------------
__global__ __launch_bounds__(256) void proj_gemm(const _Float16* __restrict__ a,
                                                 const float* __restrict__ w,
                                                 const float* __restrict__ bias,
                                                 float* __restrict__ out) {
  __shared__ _Float16 Alds[128 * 32];
  __shared__ _Float16 Blds[128 * 32];

  const int tid = threadIdx.x;
  const int wave = tid >> 5, lane = tid & 31;
  const int halfid = lane >> 4, ln16 = lane & 15;
  const int nt = blockIdx.x % 8, mt = blockIdx.x / 8;
  const int m0 = mt * 128, n0 = nt * 128;
  const int wm = wave >> 1, wn = wave & 1;

  v8f acc[2][4];
#pragma unroll
  for (int i = 0; i < 2; ++i)
#pragma unroll
    for (int j = 0; j < 4; ++j)
#pragma unroll
      for (int r = 0; r < 8; ++r) acc[i][j][r] = 0.0f;

  for (int k0 = 0; k0 < CC; k0 += 32) {
    // stage A (f16): 128 rows x 32 halves via async DMA; LDS dst is flat c*16B
#pragma unroll
    for (int i = 0; i < 2; ++i) {
      int c = tid + i * 256;                  // 0..511
      int row = c >> 2, kc8 = (c & 3) * 8;
      async_copy16(a + (size_t)(m0 + row) * CC + k0 + kc8, &Alds[row * 32 + kc8]);
    }
    // stage B (f32 -> f16): 128 rows x 32
#pragma unroll
    for (int i = 0; i < 4; ++i) {
      int id = tid + i * 256;
      int row = id >> 3, kc = (id & 7) * 4;
      const float* src = w + (size_t)(n0 + row) * CC + k0 + kc;
      float4 f = *(const float4*)src;
      v4h hh;
      hh[0] = (_Float16)f.x; hh[1] = (_Float16)f.y;
      hh[2] = (_Float16)f.z; hh[3] = (_Float16)f.w;
      *(v4h*)&Blds[row * 32 + kc] = hh;
      if (i == 0) __builtin_prefetch(src + 32, 0, 1);
    }
    wait_async_lds();
    __syncthreads();

    v16h a0 = frag16(&Alds[(wm * 32 + ln16) * 32], halfid);
    v16h a1 = frag16(&Alds[(wm * 32 + 16 + ln16) * 32], halfid);
#pragma unroll
    for (int j = 0; j < 4; ++j) {
      v16h bj = frag16(&Blds[(wn * 64 + j * 16 + ln16) * 32], halfid);
      acc[0][j] = wmma_f16(a0, bj, acc[0][j]);
      acc[1][j] = wmma_f16(a1, bj, acc[1][j]);
    }
    __syncthreads();
  }

#pragma unroll
  for (int mi = 0; mi < 2; ++mi)
#pragma unroll
    for (int ni = 0; ni < 4; ++ni) {
      int col = n0 + wn * 64 + ni * 16 + ln16;
      float bb = bias[col];
#pragma unroll
      for (int r = 0; r < 8; ++r) {
        int row_g = m0 + wm * 32 + mi * 16 + halfid * 8 + r;
        out[(size_t)row_g * CC + col] = acc[mi][ni][r] + bb;
      }
    }
}

// ---------------------------------------------------------------------------
extern "C" void kernel_launch(void* const* d_in, const int* in_sizes, int n_in,
                              void* d_out, int out_size, void* d_ws, size_t ws_size,
                              hipStream_t stream) {
  const float* x      = (const float*)d_in[0];
  const float* w_qkv  = (const float*)d_in[1];
  const float* w_proj = (const float*)d_in[2];
  const float* b_proj = (const float*)d_in[3];
  const float* fcos   = (const float*)d_in[4];
  const float* fsin   = (const float*)d_in[5];

  char* ws = (char*)d_ws;
  const size_t F32SZ = (size_t)BB * HH * NN * DD * 4;   // 16 MB
  const size_t F16SZ = (size_t)BB * HH * NN * DD * 2;   // 8 MB
  float*    qf    = (float*)(ws);
  float*    kf    = (float*)(ws + F32SZ);
  float*    vf    = (float*)(ws + 2 * F32SZ);
  _Float16* qh    = (_Float16*)(ws + 3 * F32SZ);
  _Float16* kh    = (_Float16*)(ws + 3 * F32SZ + F16SZ);
  _Float16* vh    = (_Float16*)(ws + 3 * F32SZ + 2 * F16SZ);
  _Float16* attnw = (_Float16*)(ws + 3 * F32SZ + 3 * F16SZ);
  float* out = (float*)d_out;

  qkv_gemm<<<dim3((4096 / 128) * (3 * CC / 128)), dim3(256), 0, stream>>>(x, w_qkv, qf, kf, vf);
  lnrope<<<dim3((BB * HH * NN) / 8), dim3(256), 0, stream>>>(qf, kf, vf, fcos, fsin, qh, kh, vh);
  attn<<<dim3(BB * HH * (NN / 128)), dim3(256), 0, stream>>>(qh, kh, vh, attnw);
  proj_gemm<<<dim3((4096 / 128) * (CC / 128)), dim3(256), 0, stream>>>(attnw, w_proj, b_proj, out);
}
